// MHSA_9732395893115
// MI455X (gfx1250) — compile-verified, matrix operand order checked by reading, not run on hardware
//
#include <hip/hip_runtime.h>
#include <math.h>

// Problem constants (from reference)
#define Bd    16
#define Cd    512
#define HEADd 8
#define Dd    64
#define Nd    1024
#define Wd    32
#define Hd    32

typedef __bf16 bf16_t;
typedef bf16_t v16bf __attribute__((ext_vector_type(16)));
typedef bf16_t v8bf  __attribute__((ext_vector_type(8)));
typedef float  v8f   __attribute__((ext_vector_type(8)));

union V16U { v16bf v; v8bf h[2]; bf16_t e[16]; };

// Load one WMMA 16-bit operand's per-lane data: two contiguous 16-byte chunks.
// For both A (16x32, M=row) and B (32x16, N=col) bf16 operands, lane l holds
// row/col (l%16) with K chunks {half*8 .. half*8+7} and {16+half*8 .. +7},
// half = l>>4. Caller passes the two element offsets.
__device__ __forceinline__ v16bf load_op(const bf16_t* __restrict__ base,
                                         int off0, int off1) {
  V16U u;
  u.h[0] = *(const v8bf*)(base + off0);
  u.h[1] = *(const v8bf*)(base + off1);
  return u.v;
}

__device__ __forceinline__ v8f wmma_bf16(v16bf a, v16bf b, v8f c) {
  return __builtin_amdgcn_wmma_f32_16x16x32_bf16(false, a, false, b,
                                                 (short)0, c, false, false);
}

// ---------------- prep kernels (fp32 -> bf16, transposes, pos) -------------

__global__ void prep_xT(const float* __restrict__ x, bf16_t* __restrict__ xT) {
  int i = blockIdx.x * 256 + threadIdx.x;
  if (i >= Bd * Cd * Nd) return;
  int n = i & (Nd - 1);
  int c = (i >> 10) & (Cd - 1);
  int b = i >> 19;
  xT[((size_t)b * Nd + n) * Cd + c] = (bf16_t)x[i];
}

__global__ void prep_w(const float* __restrict__ Wq, const float* __restrict__ Wk,
                       const float* __restrict__ Wv, bf16_t* __restrict__ Wb) {
  int i = blockIdx.x * 256 + threadIdx.x;
  if (i >= Cd * Cd) return;
  Wb[i]               = (bf16_t)Wq[i];
  Wb[Cd * Cd + i]     = (bf16_t)Wk[i];
  Wb[2 * Cd * Cd + i] = (bf16_t)Wv[i];
}

__global__ void prep_pos(const float* __restrict__ rel_h,
                         const float* __restrict__ rel_w,
                         bf16_t* __restrict__ PosT) {
  int i = blockIdx.x * 256 + threadIdx.x;
  if (i >= HEADd * Nd * Dd) return;
  int d = i & (Dd - 1);
  int n = (i >> 6) & (Nd - 1);
  int h = i >> 16;
  int hh = n & (Hd - 1);
  int ww = n >> 5;
  float v = rel_h[(h * Dd + d) * Hd + hh] + rel_w[(h * Dd + d) * Wd + ww];
  PosT[i] = (bf16_t)v;  // PosT[h][n][d]
}

// ---------------- projection GEMM: q/k/v = W x + b -------------------------
// One wave computes one 16(o) x 16(n) output tile, K=512 in 16 WMMA steps.
// Outputs: QT/KT as [b,h][n][64] (row n contiguous in d), V as [b,h][64][n].
__global__ void proj_kernel(const bf16_t* __restrict__ Wb,
                            const bf16_t* __restrict__ xT,
                            const float* __restrict__ bq,
                            const float* __restrict__ bk,
                            const float* __restrict__ bv,
                            bf16_t* __restrict__ QT, bf16_t* __restrict__ KT,
                            bf16_t* __restrict__ Vv) {
  int wave = threadIdx.x >> 5;
  int lane = threadIdx.x & 31;
  int task = blockIdx.x * 8 + wave;  // 3*16*32*64 tasks total
  int nt = task & 63;
  int ot = (task >> 6) & 31;
  int b  = (task >> 11) & 15;
  int proj = task >> 15;  // 0=q 1=k 2=v
  int half = lane >> 4;
  int lr   = lane & 15;
  int o0 = ot * 16, n0 = nt * 16;

  const bf16_t* Wp = Wb + (size_t)proj * Cd * Cd;
  const float* bias = (proj == 0) ? bq : ((proj == 1) ? bk : bv);

  v8f acc;
#pragma unroll
  for (int r = 0; r < 8; ++r) acc[r] = bias[o0 + r + 8 * half];

  const bf16_t* wrow = Wp + (size_t)(o0 + lr) * Cd;             // A: row o
  const bf16_t* xrow = xT + ((size_t)b * Nd + n0 + lr) * Cd;    // B: col n

#pragma unroll 4
  for (int c0 = 0; c0 < Cd; c0 += 32) {
    v16bf a = load_op(wrow, c0 + half * 8, c0 + 16 + half * 8);
    v16bf bm = load_op(xrow, c0 + half * 8, c0 + 16 + half * 8);
    acc = wmma_bf16(a, bm, acc);
  }

  if (proj == 2) {
#pragma unroll
    for (int r = 0; r < 8; ++r) {
      int o = o0 + r + 8 * half;
      int bh = b * HEADd + (o >> 6);
      int dd = o & 63;
      Vv[((size_t)bh * Dd + dd) * Nd + n0 + lr] = (bf16_t)acc[r];
    }
  } else {
    bf16_t* dst = (proj == 0) ? QT : KT;
#pragma unroll
    for (int r = 0; r < 8; ++r) {
      int o = o0 + r + 8 * half;
      int bh = b * HEADd + (o >> 6);
      int dd = o & 63;
      dst[((size_t)bh * Nd + n0 + lr) * Dd + dd] = (bf16_t)acc[r];
    }
  }
}

// ---------------- fused attention (register-blocked: 2 strips/wave) --------
// Per wave: one (b,h) and TWO 16-wide column strips (n0, n0+16) of S^T.
// S^T[m,n] = sum_d k[d,m]q[d,n] + sum_d q[d,m]pos[d,n]  (K=128 stacked GEMM)
// All A-operand loads (K rows, Q rows, V rows) are reused across both strips,
// doubling arithmetic intensity of the inner loop (24 WMMA : 24 b128 loads).
// Online softmax over m is within-lane plus one lane^16 exchange per strip;
// P^T C/D regs map lane-for-lane onto the PV WMMA B-operand layout (no LDS).
__global__ void attn_kernel(const bf16_t* __restrict__ QT,
                            const bf16_t* __restrict__ KT,
                            const bf16_t* __restrict__ Vv,
                            const bf16_t* __restrict__ PosT,
                            const float* __restrict__ x,
                            float* __restrict__ out) {
  int wave = threadIdx.x >> 5;
  int lane = threadIdx.x & 31;
  int task = blockIdx.x * 8 + wave;  // 16*8*32 tasks
  int nt = task & 31;                // 32 double-strips
  int h  = (task >> 5) & 7;
  int b  = task >> 8;
  int half = lane >> 4;
  int lr   = lane & 15;
  int n0 = nt * 32;
  int bh = b * HEADd + h;

  // Preload B operands for S^T GEMM: 2 strips x (q seg0/1, pos seg0/1)
  v16bf Bq[2][2], Bp[2][2];
#pragma unroll
  for (int s = 0; s < 2; ++s) {
    const bf16_t* qrow = QT + ((size_t)bh * Nd + n0 + s * 16 + lr) * Dd;
    const bf16_t* prow = PosT + ((size_t)h * Nd + n0 + s * 16 + lr) * Dd;
    Bq[s][0] = load_op(qrow, half * 8,      16 + half * 8);
    Bq[s][1] = load_op(qrow, 32 + half * 8, 48 + half * 8);
    Bp[s][0] = load_op(prow, half * 8,      16 + half * 8);
    Bp[s][1] = load_op(prow, 32 + half * 8, 48 + half * 8);
  }

  const bf16_t* Qbase = QT + (size_t)bh * Nd * Dd;
  const bf16_t* Kbase = KT + (size_t)bh * Nd * Dd;
  const bf16_t* Vbase = Vv + (size_t)bh * Dd * Nd;

  v8f O[2][4];
  float run_m[2], run_s[2];
#pragma unroll
  for (int s = 0; s < 2; ++s) {
#pragma unroll
    for (int t = 0; t < 4; ++t) O[s][t] = (v8f){};
    run_m[s] = -__builtin_inff();
    run_s[s] = 0.0f;
  }

  for (int m0 = 0; m0 < Nd; m0 += 32) {
    // --- shared A operands for the two 16-row m tiles ---
    const bf16_t* krow0 = Kbase + (size_t)(m0 + lr) * Dd;
    const bf16_t* qr0   = Qbase + (size_t)(m0 + lr) * Dd;
    const bf16_t* krow1 = Kbase + (size_t)(m0 + 16 + lr) * Dd;
    const bf16_t* qr1   = Qbase + (size_t)(m0 + 16 + lr) * Dd;
    v16bf ak0_0 = load_op(krow0, half * 8, 16 + half * 8);
    v16bf ak0_1 = load_op(krow0, 32 + half * 8, 48 + half * 8);
    v16bf aq0_0 = load_op(qr0,   half * 8, 16 + half * 8);
    v16bf aq0_1 = load_op(qr0,   32 + half * 8, 48 + half * 8);
    v16bf ak1_0 = load_op(krow1, half * 8, 16 + half * 8);
    v16bf ak1_1 = load_op(krow1, 32 + half * 8, 48 + half * 8);
    v16bf aq1_0 = load_op(qr1,   half * 8, 16 + half * 8);
    v16bf aq1_1 = load_op(qr1,   32 + half * 8, 48 + half * 8);

    // --- shared V A-operands (4 d-tiles) ---
    v16bf av[4];
#pragma unroll
    for (int dt = 0; dt < 4; ++dt) {
      const bf16_t* vrow = Vbase + (size_t)(dt * 16 + lr) * Nd;
      av[dt] = load_op(vrow, m0 + half * 8, m0 + 16 + half * 8);
    }

#pragma unroll
    for (int s = 0; s < 2; ++s) {
      // --- two 16-row S^T tiles for strip s (K=128: k.q + q.pos) ---
      v8f st0 = (v8f){}, st1 = (v8f){};
      st0 = wmma_bf16(ak0_0, Bq[s][0], st0);
      st0 = wmma_bf16(ak0_1, Bq[s][1], st0);
      st0 = wmma_bf16(aq0_0, Bp[s][0], st0);
      st0 = wmma_bf16(aq0_1, Bp[s][1], st0);
      st1 = wmma_bf16(ak1_0, Bq[s][0], st1);
      st1 = wmma_bf16(ak1_1, Bq[s][1], st1);
      st1 = wmma_bf16(aq1_0, Bp[s][0], st1);
      st1 = wmma_bf16(aq1_1, Bp[s][1], st1);

      // --- online softmax update (column n per lane-pair l, l^16) ---
      float lm = st0[0];
#pragma unroll
      for (int r = 0; r < 8; ++r) {
        lm = fmaxf(lm, st0[r]);
        lm = fmaxf(lm, st1[r]);
      }
      lm = fmaxf(lm, __shfl_xor(lm, 16, 32));  // sync max across lane pair
      float mnew = fmaxf(run_m[s], lm);
      float alpha = __expf(run_m[s] - mnew);
      float lsum = 0.0f;
      V16U p;
#pragma unroll
      for (int r = 0; r < 8; ++r) {
        float e0 = __expf(st0[r] - mnew);
        float e1 = __expf(st1[r] - mnew);
        lsum += e0 + e1;
        p.e[r]     = (bf16_t)e0;  // K = half*8 + r        (tile 0)
        p.e[8 + r] = (bf16_t)e1;  // K = 16 + half*8 + r   (tile 1)
      }
      run_s[s] = run_s[s] * alpha + lsum;
      run_m[s] = mnew;
#pragma unroll
      for (int t = 0; t < 4; ++t) O[s][t] = O[s][t] * alpha;

      // --- O[d,n] += V[d, m0..m0+31] * P^T ---
#pragma unroll
      for (int dt = 0; dt < 4; ++dt) O[s][dt] = wmma_bf16(av[dt], p.v, O[s][dt]);
    }
  }

  // --- normalize, residual, store ---
#pragma unroll
  for (int s = 0; s < 2; ++s) {
    float stot = run_s[s] + __shfl_xor(run_s[s], 16, 32);
    float inv = 1.0f / stot;
#pragma unroll
    for (int dt = 0; dt < 4; ++dt) {
#pragma unroll
      for (int r = 0; r < 8; ++r) {
        int ch = h * Dd + dt * 16 + r + 8 * half;
        size_t idx = ((size_t)b * Cd + ch) * Nd + n0 + s * 16 + lr;
        out[idx] = O[s][dt][r] * inv + x[idx];
      }
    }
  }
}

// ---------------- host launcher --------------------------------------------
extern "C" void kernel_launch(void* const* d_in, const int* in_sizes, int n_in,
                              void* d_out, int out_size, void* d_ws, size_t ws_size,
                              hipStream_t stream) {
  const float* x     = (const float*)d_in[0];
  const float* Wq    = (const float*)d_in[1];
  const float* bq    = (const float*)d_in[2];
  const float* Wk    = (const float*)d_in[3];
  const float* bk    = (const float*)d_in[4];
  const float* Wv    = (const float*)d_in[5];
  const float* bv    = (const float*)d_in[6];
  const float* rel_h = (const float*)d_in[7];
  const float* rel_w = (const float*)d_in[8];
  float* out = (float*)d_out;

  // workspace layout (bf16 elements)
  bf16_t* ws   = (bf16_t*)d_ws;
  const size_t nXT  = (size_t)Bd * Nd * Cd;         // 8388608
  const size_t nW   = (size_t)3 * Cd * Cd;          //  786432
  const size_t nQKV = (size_t)Bd * HEADd * Nd * Dd; // 8388608
  const size_t nPos = (size_t)HEADd * Nd * Dd;      //  524288
  bf16_t* xT   = ws;
  bf16_t* Wb   = xT + nXT;
  bf16_t* QT   = Wb + nW;
  bf16_t* KT   = QT + nQKV;
  bf16_t* Vv   = KT + nQKV;
  bf16_t* PosT = Vv + nQKV;

  // prep
  prep_xT<<<(int)((nXT + 255) / 256), 256, 0, stream>>>(x, xT);
  prep_w<<<(Cd * Cd + 255) / 256, 256, 0, stream>>>(Wq, Wk, Wv, Wb);
  prep_pos<<<(int)((nPos + 255) / 256), 256, 0, stream>>>(rel_h, rel_w, PosT);

  // projections: 3 * B * (C/16) * (N/16) wave-tasks, 8 waves per 256-thread WG
  {
    int tasks = 3 * Bd * (Cd / 16) * (Nd / 16);  // 98304
    proj_kernel<<<tasks / 8, 256, 0, stream>>>(Wb, xT, bq, bk, bv, QT, KT, Vv);
  }

  // fused attention: B * H * (N/32) wave-tasks (2 column strips per wave)
  {
    int tasks = Bd * HEADd * (Nd / 32);  // 4096
    attn_kernel<<<tasks / 8, 256, 0, stream>>>(QT, KT, Vv, PosT, x, out);
  }
}
